// ScaledDotProductAttention_6863357739210
// MI455X (gfx1250) — matrix operand order repaired and checked by reading, compiled, not run on hardware
//
#include <hip/hip_runtime.h>

typedef __attribute__((ext_vector_type(16))) _Float16 v16h;
typedef __attribute__((ext_vector_type(8)))  _Float16 v8h;
typedef __attribute__((ext_vector_type(4)))  _Float16 v4h;
typedef __attribute__((ext_vector_type(8)))  float    v8f;
typedef __attribute__((ext_vector_type(4)))  float    v4f;

#define S_LEN   2048
#define DH      64
#define ATT_SCALE 0.125f          // 1/sqrt(64)
#define MASK_VAL  -1e9f
#define QTILE   16
#define NTHREADS 256
#define CHUNK   128               // keys staged per block iteration

// Padded LDS row strides (64 banks x 4B = 256B period):
//  sS rows   : 2052 f32 (8208 B, +4 banks/row)
//  sP rows   : 2056 f16 (4112 B, +4 banks/row)
//  sStage row: 136 f16  (272 B = 68 dwords = +4 banks/row -> b128 reads conflict-free)
#define SS_STRIDE 2052
#define SP_STRIDE 2056
#define ST_STRIDE 136
#define SS_QSTRIDE (SS_STRIDE/4)
#define SP_QSTRIDE (SP_STRIDE/4)

#define SS_BYTES   (16*SS_STRIDE*4)        // 131328
#define SP_BYTES   (16*SP_STRIDE*2)        //  65792
#define ST_BYTES   (CHUNK*ST_STRIDE*2)     //  34816 (shared by K-stage and V^T-stage)
#define SMEM_BYTES (SS_BYTES + SP_BYTES + ST_BYTES + (256 + 32 + 2048)*4)

__global__ __launch_bounds__(NTHREADS)
void ScaledDotProductAttention_6863357739210_kernel(
    const float* __restrict__ Q, const float* __restrict__ K,
    const float* __restrict__ V, const unsigned char* __restrict__ mask,
    float* __restrict__ ctx_out, float* __restrict__ prob_out)
{
    extern __shared__ char smem_raw[];
    float*    sS     = (float*)smem_raw;
    _Float16* sP     = (_Float16*)(smem_raw + SS_BYTES);
    _Float16* sStage = (_Float16*)(smem_raw + SS_BYTES + SP_BYTES);
    float*    sRed   = (float*)(smem_raw + SS_BYTES + SP_BYTES + ST_BYTES);
    float*    sStat  = sRed + 256;
    float*    sCtx   = sStat + 32;
    v4f*      sSq    = (v4f*)sS;
    v4h*      sPq    = (v4h*)sP;

    const int tid  = threadIdx.x;
    const int lane = tid & 31;
    const int wave = tid >> 5;
    const int bh   = blockIdx.y;          // b*H + h
    const int q0   = blockIdx.x * QTILE;  // first query row of this tile

    const float* Qb = Q + ((size_t)bh * S_LEN + q0) * DH;
    const float* Kb = K + (size_t)bh * S_LEN * DH;
    const float* Vb = V + (size_t)bh * S_LEN * DH;
    const unsigned char* Mb = mask + ((size_t)bh * S_LEN + q0) * (size_t)S_LEN;

    const int hrow = lane & 15;   // A: matrix row / B: matrix column
    const int kg   = lane >> 4;   // lane-half selecting the K sub-range

    // ---------------- load Q A-fragments (16x32 f16, d 0..31 and 32..63) ----
    v16h aQ0, aQ1;
    {
        const float* qrow = Qb + hrow * DH;
        #pragma unroll
        for (int v = 0; v < 8; ++v) {
            const int kd = 2*v + 16*kg;
            aQ0[2*v]   = (_Float16)qrow[kd];
            aQ0[2*v+1] = (_Float16)qrow[kd + 1];
            aQ1[2*v]   = (_Float16)qrow[32 + kd];
            aQ1[2*v+1] = (_Float16)qrow[32 + kd + 1];
        }
    }

    // ---------------- Pass 1: scores = scale * Q K^T -> LDS ----------------
    // K staged per 128-key chunk as f16 row-major (coalesced b128 global loads,
    // then contiguous 32B LDS reads for B-fragments).
    for (int c = 0; c < 16; ++c) {
        const int kb = c * CHUNK;
        {
            const int qcol = (tid & 15) * 4;   // 16 threads cover one 256B row
            const int kr0  = tid >> 4;
            #pragma unroll
            for (int j = 0; j < 8; ++j) {
                const int kr = kr0 + 16*j;
                const v4f kq = *(const v4f*)(Kb + (size_t)(kb + kr) * DH + qcol);
                v4h kh;
                kh[0] = (_Float16)kq[0]; kh[1] = (_Float16)kq[1];
                kh[2] = (_Float16)kq[2]; kh[3] = (_Float16)kq[3];
                *(v4h*)(sStage + kr * ST_STRIDE + qcol) = kh;
            }
        }
        __syncthreads();
        {
            const _Float16* krow = sStage + (wave*16 + hrow) * ST_STRIDE;
            const v16h b0 = *(const v16h*)(krow + kg*16);        // dims 0..31
            const v16h b1 = *(const v16h*)(krow + 32 + kg*16);   // dims 32..63
            v8f cc = {};
            cc = __builtin_amdgcn_wmma_f32_16x16x32_f16(false, aQ0, false, b0,
                                                        (short)0, cc, false, false);
            cc = __builtin_amdgcn_wmma_f32_16x16x32_f16(false, aQ1, false, b1,
                                                        (short)0, cc, false, false);
            #pragma unroll
            for (int r = 0; r < 8; ++r) {
                const int m = r + 8*kg;
                sS[m * SS_STRIDE + kb + wave*16 + hrow] = cc[r] * ATT_SCALE;
            }
        }
        __syncthreads();
    }

    // ---------------- softmax: mask + row max (float4 sweeps) --------------
    const int srow = tid >> 4;    // 16 rows, 16 threads each
    const int sj   = tid & 15;    // quad lane within row
    {
        float lmax = -3.402823e38f;
        for (int i = 0; i < 32; ++i) {
            const int q = sj + 16*i;                  // quad index, 0..511
            v4f s = sSq[srow * SS_QSTRIDE + q];
            const unsigned int mw =
                *(const unsigned int*)(Mb + (size_t)srow * S_LEN + q*4);
            if (mw) {
                if (mw & 0x000000FFu) s.x = MASK_VAL;
                if (mw & 0x0000FF00u) s.y = MASK_VAL;
                if (mw & 0x00FF0000u) s.z = MASK_VAL;
                if (mw & 0xFF000000u) s.w = MASK_VAL;
                sSq[srow * SS_QSTRIDE + q] = s;
            }
            lmax = fmaxf(lmax, fmaxf(fmaxf(s.x, s.y), fmaxf(s.z, s.w)));
        }
        sRed[srow*16 + sj] = lmax;
    }
    __syncthreads();
    if (tid < 16) {
        float m = sRed[tid*16];
        for (int j = 1; j < 16; ++j) m = fmaxf(m, sRed[tid*16 + j]);
        sStat[tid] = m;
    }
    __syncthreads();

    // ---------------- exp + row sum, store P as f16 ------------------------
    {
        const float rmax = sStat[srow];
        float lsum = 0.0f;
        for (int i = 0; i < 32; ++i) {
            const int q = sj + 16*i;
            const v4f s = sSq[srow * SS_QSTRIDE + q];
            const float p0 = __expf(s.x - rmax);
            const float p1 = __expf(s.y - rmax);
            const float p2 = __expf(s.z - rmax);
            const float p3 = __expf(s.w - rmax);
            v4h ph; ph[0] = (_Float16)p0; ph[1] = (_Float16)p1;
                    ph[2] = (_Float16)p2; ph[3] = (_Float16)p3;
            sPq[srow * SP_QSTRIDE + q] = ph;
            lsum += (p0 + p1) + (p2 + p3);
        }
        sRed[srow*16 + sj] = lsum;
    }
    __syncthreads();
    if (tid < 16) {
        float s = 0.0f;
        for (int j = 0; j < 16; ++j) s += sRed[tid*16 + j];
        sStat[16 + tid] = 1.0f / s;
    }
    __syncthreads();

    // ---------------- write attn_prob (coalesced float4 stores) ------------
    {
        float* prob_base = prob_out + ((size_t)bh * S_LEN + q0) * (size_t)S_LEN;
        for (int r = 0; r < 16; ++r) {
            const float inv = sStat[16 + r];
            v4f* prow = (v4f*)(prob_base + (size_t)r * S_LEN);
            #pragma unroll
            for (int it = 0; it < 2; ++it) {
                const int q = tid + it * NTHREADS;    // quad index, 0..511
                const v4h ph = sPq[r * SP_QSTRIDE + q];
                v4f o;
                o.x = (float)ph[0] * inv; o.y = (float)ph[1] * inv;
                o.z = (float)ph[2] * inv; o.w = (float)ph[3] * inv;
                prow[q] = o;
            }
        }
    }

    // ---------------- Pass 2: context = P V --------------------------------
    // V staged per 128-key chunk as f16 transposed (sVt[d][k]) via a register
    // 4x8 micro-transpose; B-fragments become contiguous 32B LDS reads.
    const int dtile = wave & 3;   // 16-wide slice of D
    const int khalf = wave >> 2;  // 64-key half of each chunk
    v8f cacc = {};
    for (int c = 0; c < 16; ++c) {
        const int kb = c * CHUNK;
        __syncthreads();          // previous chunk fully consumed
        {
            const int d0 = (tid & 15) * 4;
            const int k0 = (tid >> 4) * 8;
            v4f rv[8];
            #pragma unroll
            for (int j = 0; j < 8; ++j)
                rv[j] = *(const v4f*)(Vb + (size_t)(kb + k0 + j) * DH + d0);
            #pragma unroll
            for (int i = 0; i < 4; ++i) {
                v8h oh;
                #pragma unroll
                for (int j = 0; j < 8; ++j) oh[j] = (_Float16)rv[j][i];
                *(v8h*)(sStage + (d0 + i) * ST_STRIDE + k0) = oh;
            }
        }
        __syncthreads();
        #pragma unroll
        for (int sub = 0; sub < 2; ++sub) {
            const int kloc = khalf*64 + sub*32;       // key offset within chunk
            v16h ap;
            #pragma unroll
            for (int v = 0; v < 8; ++v) {
                const int kk = 2*v + 16*kg;
                ap[2*v]   = sP[hrow * SP_STRIDE + kb + kloc + kk];
                ap[2*v+1] = sP[hrow * SP_STRIDE + kb + kloc + kk + 1];
            }
            const v16h bv = *(const v16h*)(sStage + (dtile*16 + hrow) * ST_STRIDE
                                           + kloc + kg*16);
            cacc = __builtin_amdgcn_wmma_f32_16x16x32_f16(false, ap, false, bv,
                                                          (short)0, cacc, false, false);
        }
    }
    #pragma unroll
    for (int r = 0; r < 8; ++r)
        sCtx[(wave*8 + r)*32 + lane] = cacc[r];
    __syncthreads();

    // combine the two k-halves, apply 1/rowsum, write context
    {
        float* ctx_base = ctx_out + ((size_t)bh * S_LEN + q0) * DH;
        for (int e = tid; e < 1024; e += NTHREADS) {
            const int dt  = e >> 8;
            const int rem = e & 255;
            const int m   = rem >> 4;
            const int n   = rem & 15;
            const int l   = n + ((m >> 3) << 4);
            const int rr  = m & 7;
            const float v0 = sCtx[((dt    )*8 + rr)*32 + l];
            const float v1 = sCtx[((dt + 4)*8 + rr)*32 + l];
            ctx_base[(size_t)m * DH + dt*16 + n] = (v0 + v1) * sStat[16 + m];
        }
    }
}

extern "C" void kernel_launch(void* const* d_in, const int* in_sizes, int n_in,
                              void* d_out, int out_size, void* d_ws, size_t ws_size,
                              hipStream_t stream) {
    (void)in_sizes; (void)n_in; (void)d_ws; (void)ws_size; (void)out_size;
    const float* Q = (const float*)d_in[0];
    const float* K = (const float*)d_in[1];
    const float* V = (const float*)d_in[2];
    const unsigned char* mask = (const unsigned char*)d_in[3]; // jnp bool = 1 byte

    const size_t B = 4, H = 16;
    float* ctx  = (float*)d_out;
    float* prob = (float*)d_out + B * H * (size_t)S_LEN * DH;  // (context, attn_prob) concat

    // Allow >64KB dynamic LDS (CDNA5 WGP has 320KB)
    (void)hipFuncSetAttribute(
        (const void*)ScaledDotProductAttention_6863357739210_kernel,
        hipFuncAttributeMaxDynamicSharedMemorySize, (int)SMEM_BYTES);

    dim3 grid(S_LEN / QTILE, (unsigned)(B * H));
    ScaledDotProductAttention_6863357739210_kernel
        <<<grid, NTHREADS, SMEM_BYTES, stream>>>(Q, K, V, mask, ctx, prob);
}
